// ScaledLowRankElmanCell_16303695856122
// MI455X (gfx1250) — matrix operand order repaired and checked by reading, not compiled, not run on hardware
//
#include <hip/hip_runtime.h>
#include <hip/hip_bf16.h>
#include <stdint.h>

// Problem sizes (fixed by the reference)
#define T_STEPS 1024
#define BATCH   32
#define DIM     1024
#define RANK    128
#define LDPAD   8   // bf16 elements of row padding (keeps 16B alignment, skews banks)

typedef __attribute__((ext_vector_type(16))) __bf16 bf16x16;
typedef __attribute__((ext_vector_type(8)))  __bf16 bf16x8;
typedef __attribute__((ext_vector_type(8)))  float  f32x8;
typedef __attribute__((ext_vector_type(4)))  float  f32x4;
typedef unsigned int u32x4 __attribute__((ext_vector_type(4)));
typedef int          i32x4 __attribute__((ext_vector_type(4)));
typedef int          i32x8 __attribute__((ext_vector_type(8)));

// Native RNE float -> bf16 (lowers to hardware cvt on gfx1250)
__device__ __forceinline__ __bf16 f2bf(float f) { return (__bf16)f; }

// CDNA5 16-bit A-fragment (16x32): per lane two contiguous 8-half chunks,
// chunk base offsets differ by 16 K-elements (hi-lane K skew handled by caller).
__device__ __forceinline__ bf16x16 make_afrag(const __bf16* p0, const __bf16* p1) {
  bf16x8 lo = *(const bf16x8*)p0;
  bf16x8 hi = *(const bf16x8*)p1;
  return __builtin_shufflevector(lo, hi, 0,1,2,3,4,5,6,7,8,9,10,11,12,13,14,15);
}

__device__ __forceinline__ f32x8 wmma_bf16(bf16x16 a, bf16x16 b, f32x8 c) {
  return __builtin_amdgcn_wmma_f32_16x16x32_bf16(
      /*neg_a=*/false, a, /*neg_b=*/false, b,
      /*c_mod=*/(short)0, c, /*reuse_a=*/false, /*reuse_b=*/false);
}

// ---------------------------------------------------------------------------
// TDM: DMA a 1-D fp32 vector (len floats) from global memory into LDS using a
// Tensor DMA Descriptor (D#).  Group0: count=1, lds_addr, global_addr, type=2.
// Group1: data_size=4B, tensor_dim0=tile_dim0=len, tensor_dim1=tile_dim1=1.
// Tracked with TENSORcnt; caller barriers afterwards.
// ---------------------------------------------------------------------------
__device__ __forceinline__ void tdm_load_f32_vec(const float* gsrc,
                                                 uint32_t lds_byte_off,
                                                 uint32_t len) {
#if __has_builtin(__builtin_amdgcn_tensor_load_to_lds)
  const uint64_t ga = (uint64_t)(uintptr_t)gsrc;
  u32x4 g0;
  g0[0] = 1u;                                    // count=1, user descriptor
  g0[1] = lds_byte_off;                          // D#.lds_addr
  g0[2] = (uint32_t)ga;                          // D#.global_addr[31:0]
  g0[3] = (uint32_t)((ga >> 32) & 0x01FFFFFFu)   // D#.global_addr[56:32]
        | 0x80000000u;                           // D#.type = 2 ("image")
  i32x8 g1;
  g1[0] = (int)(2u << 16);                       // data_size = 4 bytes
  g1[1] = (int)((len & 0xFFFFu) << 16);          // tensor_dim0[15:0]
  g1[2] = (int)((len >> 16) | (1u << 16));       // tensor_dim0[31:16] | tensor_dim1=1
  g1[3] = (int)((len & 0xFFFFu) << 16);          // tile_dim0 = len
  g1[4] = 1;                                     // tile_dim1 = 1, tile_dim2 = 0
  g1[5] = (int)len;                              // tensor_dim0_stride[31:0]
  g1[6] = 0;                                     // stride0[47:32] | stride1[15:0]
  g1[7] = 0;
  i32x4 gz = {0, 0, 0, 0};
#if __clang_major__ >= 23
  i32x8 gz8 = {0, 0, 0, 0, 0, 0, 0, 0};
  __builtin_amdgcn_tensor_load_to_lds(g0, g1, gz, gz, gz8, 0);
#else
  __builtin_amdgcn_tensor_load_to_lds(g0, g1, gz, gz, 0);
#endif
#if __has_builtin(__builtin_amdgcn_s_wait_tensorcnt)
  __builtin_amdgcn_s_wait_tensorcnt(0);
#endif
#else
  (void)gsrc; (void)lds_byte_off; (void)len;
#endif
}

// ---------------------------------------------------------------------------
// Kernel 0: convert weights to bf16 in workspace, folding the diagonal scales
// into the U matrices:  Ux' = U_x * s_x,  Uh' = U_h * s_h.
// Layouts chosen so WMMA B-fragments are K-contiguous per lane:
//   W_V* : [R, D] row-major (B[k=d, n=r] = V[r, d], row r contiguous over d)
//   W_U* : [D, R] row-major (B[k=r, n=d] = U'[d, r], row d contiguous over r)
// ---------------------------------------------------------------------------
__global__ __launch_bounds__(256) void convert_weights_kernel(
    const float* __restrict__ U_h, const float* __restrict__ V_h,
    const float* __restrict__ s_h, const float* __restrict__ U_x,
    const float* __restrict__ V_x, const float* __restrict__ s_x,
    __bf16* __restrict__ W_Vx, __bf16* __restrict__ W_Ux,
    __bf16* __restrict__ W_Vh, __bf16* __restrict__ W_Uh) {
  int i = blockIdx.x * blockDim.x + threadIdx.x;   // grid covers R*D == D*R
  if (i < RANK * DIM) {
    W_Vx[i] = f2bf(V_x[i]);
    W_Vh[i] = f2bf(V_h[i]);
    int r = i & (RANK - 1);                        // U mats are [D, R]
    W_Ux[i] = f2bf(U_x[i] * s_x[r]);
    W_Uh[i] = f2bf(U_h[i] * s_h[r]);
  }
}

// ---------------------------------------------------------------------------
// Kernel 1: input projection, parallel over T.
//   Vx  = x_t @ W_Vx^T            [32,128]
//   Ux  = Vx  @ W_Ux^T            [32,1024]  (scale already folded in)
// Ux_t is written fp32 into the h-output region of d_out at slot t+1; the
// scan kernel consumes it there and overwrites it in place with h_t.
// One 256-thread block (8 wave32 waves) per timestep.
// ---------------------------------------------------------------------------
__global__ __launch_bounds__(256) void input_proj_kernel(
    const float* __restrict__ x,
    const __bf16* __restrict__ W_Vx, const __bf16* __restrict__ W_Ux,
    float* __restrict__ ux_slots /* points at h-region slot 1 */) {
  __shared__ __align__(64) __bf16 vxbuf[BATCH][RANK + LDPAD];

  const int t    = blockIdx.x;
  const int lane = threadIdx.x & 31;
  const int wave = threadIdx.x >> 5;   // 0..7
  const int n_l  = lane & 15;          // N (or M for A-frags) within tile
  const int hi   = lane >> 4;          // lane-half selects K skew

  const float* xt = x + (size_t)t * BATCH * DIM;

  // ---- GEMM1: wave w owns Vx column tile nt = w; K = DIM in 32-chunks ----
  for (int mt = 0; mt < 2; ++mt) {
    f32x8 acc = {};
    const __bf16* brow  = W_Vx + (size_t)(wave * 16 + n_l) * DIM + hi * 8;
    const float*  arow  = xt + (size_t)(mt * 16 + n_l) * DIM + hi * 8;
    for (int kt = 0; kt < DIM / 32; ++kt) {
      // A fragment straight from global fp32 x_t (converted in-register)
      const float* pa = arow + kt * 32;
      f32x4 a0 = *(const f32x4*)(pa);
      f32x4 a1 = *(const f32x4*)(pa + 4);
      f32x4 a2 = *(const f32x4*)(pa + 16);
      f32x4 a3 = *(const f32x4*)(pa + 20);
      bf16x16 af;
#pragma unroll
      for (int e = 0; e < 4; ++e) {
        af[e]      = f2bf(a0[e]);
        af[4 + e]  = f2bf(a1[e]);
        af[8 + e]  = f2bf(a2[e]);
        af[12 + e] = f2bf(a3[e]);
      }
      bf16x16 bfr = make_afrag(brow + kt * 32, brow + kt * 32 + 16);
      __builtin_prefetch(brow + (kt + 1) * 32, 0, 0);   // global_prefetch_b8
      acc = wmma_bf16(af, bfr, acc);
    }
    // C layout: VGPR v -> M = v + 8*hi, N = n_l
#pragma unroll
    for (int v = 0; v < 8; ++v)
      vxbuf[mt * 16 + hi * 8 + v][wave * 16 + n_l] = f2bf(acc[v]);
  }
  __syncthreads();

  // ---- GEMM2: wave w owns d in [w*128, (w+1)*128); K = RANK in 4 chunks ----
  float* dst = ux_slots + (size_t)t * BATCH * DIM;
  for (int mt = 0; mt < 2; ++mt) {
    bf16x16 af[4];
#pragma unroll
    for (int kt = 0; kt < 4; ++kt) {
      const __bf16* p = &vxbuf[mt * 16 + n_l][kt * 32 + hi * 8];
      af[kt] = make_afrag(p, p + 16);
    }
    for (int j = 0; j < 8; ++j) {
      const int nt = wave * 8 + j;
      f32x8 acc = {};
      const __bf16* brow = W_Ux + (size_t)(nt * 16 + n_l) * RANK + hi * 8;
#pragma unroll
      for (int kt = 0; kt < 4; ++kt) {
        bf16x16 bfr = make_afrag(brow + kt * 32, brow + kt * 32 + 16);
        acc = wmma_bf16(af[kt], bfr, acc);
      }
#pragma unroll
      for (int v = 0; v < 8; ++v)
        dst[(size_t)(mt * 16 + hi * 8 + v) * DIM + nt * 16 + n_l] = acc[v];
    }
  }
}

// ---------------------------------------------------------------------------
// Kernel 2: the serial scan. One persistent 256-thread workgroup.
//   Vh  = h   @ W_Vh^T   [32,128]   (h held bf16 in LDS)
//   pre = Vh  @ W_Uh^T   [32,1024]  (scale folded)
//   h   = tanh(pre + Ux_t + b);  y_t = h * silu(z_t)
// Ux_t is read from h-slot t+1 of d_out and overwritten in place with h_t.
// Weights stream from L2 (256 KB each, fully L2-resident, reused 1024x).
// Bias is DMA'd into LDS by the Tensor Data Mover.
// ---------------------------------------------------------------------------
__global__ __launch_bounds__(256) void scan_kernel(
    const float* __restrict__ z, const float* __restrict__ h0,
    const __bf16* __restrict__ W_Vh, const __bf16* __restrict__ W_Uh,
    const float* __restrict__ bias, float* __restrict__ out) {
  __shared__ __align__(64) __bf16 hbuf[BATCH][DIM + LDPAD];
  __shared__ __align__(64) __bf16 vhbuf[BATCH][RANK + LDPAD];
  __shared__ __align__(64) float bias_s[DIM];

  float* out_y = out;                                    // [T, B, D]
  float* out_h = out + (size_t)T_STEPS * BATCH * DIM;    // [T+1, B, D]

  const int tid  = threadIdx.x;
  const int lane = tid & 31;
  const int wave = tid >> 5;
  const int n_l  = lane & 15;
  const int hi   = lane >> 4;

  // Prologue: h0 -> h-slot 0 (fp32) and hbuf (bf16); bias -> LDS via TDM.
#if __has_builtin(__builtin_amdgcn_tensor_load_to_lds)
  if (wave == 0)   // wave-uniform branch; TDM ignores EXEC, issue once
    tdm_load_f32_vec(bias, (uint32_t)(uintptr_t)&bias_s[0], DIM);
#else
  for (int i = tid; i < DIM; i += 256) bias_s[i] = bias[i];
#endif
  for (int i = tid; i < BATCH * DIM; i += 256) {
    float v = h0[i];
    out_h[i] = v;
    hbuf[i >> 10][i & (DIM - 1)] = f2bf(v);
  }
#if __has_builtin(__builtin_amdgcn_s_cluster_barrier)
  __builtin_amdgcn_s_cluster_barrier();   // NOP when not dispatched as cluster
#endif
  __syncthreads();

  for (int t = 0; t < T_STEPS; ++t) {
    // ---- Phase A: Vh = h @ W_Vh^T (wave w owns rank tile nt = w) ----
    for (int mt = 0; mt < 2; ++mt) {
      f32x8 acc = {};
      const __bf16* brow = W_Vh + (size_t)(wave * 16 + n_l) * DIM + hi * 8;
      const __bf16* arow = &hbuf[mt * 16 + n_l][hi * 8];
      for (int kt = 0; kt < DIM / 32; ++kt) {
        bf16x16 af  = make_afrag(arow + kt * 32, arow + kt * 32 + 16);
        bf16x16 bfr = make_afrag(brow + kt * 32, brow + kt * 32 + 16);
        __builtin_prefetch(brow + (kt + 1) * 32, 0, 0);
        acc = wmma_bf16(af, bfr, acc);
      }
#pragma unroll
      for (int v = 0; v < 8; ++v)
        vhbuf[mt * 16 + hi * 8 + v][wave * 16 + n_l] = f2bf(acc[v]);
    }
    __syncthreads();

    // ---- Phase B + fused epilogue (wave w owns d in [w*128,(w+1)*128)) ----
    float* hslot = out_h + (size_t)(t + 1) * BATCH * DIM;  // Ux_t in, h_t out
    const float* zt = z + (size_t)t * BATCH * DIM;
    float* yt = out_y + (size_t)t * BATCH * DIM;

    for (int mt = 0; mt < 2; ++mt) {
      bf16x16 af[4];
#pragma unroll
      for (int kt = 0; kt < 4; ++kt) {
        const __bf16* p = &vhbuf[mt * 16 + n_l][kt * 32 + hi * 8];
        af[kt] = make_afrag(p, p + 16);
      }
      for (int j = 0; j < 8; ++j) {
        const int nt = wave * 8 + j;
        f32x8 acc = {};
        const __bf16* brow = W_Uh + (size_t)(nt * 16 + n_l) * RANK + hi * 8;
#pragma unroll
        for (int kt = 0; kt < 4; ++kt) {
          bf16x16 bfr = make_afrag(brow + kt * 32, brow + kt * 32 + 16);
          acc = wmma_bf16(af[kt], bfr, acc);
        }
#pragma unroll
        for (int v = 0; v < 8; ++v) {
          const int m = mt * 16 + hi * 8 + v;      // batch row
          const int d = nt * 16 + n_l;             // feature column
          const size_t idx = (size_t)m * DIM + d;
          float hnew = tanhf(acc[v] + hslot[idx] + bias_s[d]);
          hslot[idx] = hnew;                       // overwrite Ux_t with h_t
          hbuf[m][d] = f2bf(hnew);                 // refresh bf16 state
          float zv  = zt[idx];
          float sig = 1.0f / (1.0f + __expf(-zv));
          yt[idx]   = hnew * (zv * sig);           // h * silu(z)
        }
      }
    }
    __syncthreads();   // hbuf/vhbuf consistent before next step's Phase A
  }
}

extern "C" void kernel_launch(void* const* d_in, const int* in_sizes, int n_in,
                              void* d_out, int out_size, void* d_ws, size_t ws_size,
                              hipStream_t stream) {
  (void)in_sizes; (void)n_in; (void)out_size; (void)ws_size;

  const float* x   = (const float*)d_in[0];
  const float* z   = (const float*)d_in[1];
  const float* h0  = (const float*)d_in[2];
  const float* U_h = (const float*)d_in[3];
  const float* V_h = (const float*)d_in[4];
  const float* s_h = (const float*)d_in[5];
  const float* U_x = (const float*)d_in[6];
  const float* V_x = (const float*)d_in[7];
  const float* s_x = (const float*)d_in[8];
  const float* b   = (const float*)d_in[9];
  float* out = (float*)d_out;

  // Workspace: 4 bf16 weight matrices, 256 KB each (1 MB total).
  __bf16* W_Vx = (__bf16*)d_ws;
  __bf16* W_Ux = (__bf16*)((char*)d_ws + 262144);
  __bf16* W_Vh = (__bf16*)((char*)d_ws + 524288);
  __bf16* W_Uh = (__bf16*)((char*)d_ws + 786432);

  convert_weights_kernel<<<(RANK * DIM) / 256, 256, 0, stream>>>(
      U_h, V_h, s_h, U_x, V_x, s_x, W_Vx, W_Ux, W_Vh, W_Uh);

  // Ux_t goes into h-output slot t+1 (in place, consumed by the scan).
  float* ux_slots = out + (size_t)T_STEPS * BATCH * DIM + (size_t)BATCH * DIM;
  input_proj_kernel<<<T_STEPS, 256, 0, stream>>>(x, W_Vx, W_Ux, ux_slots);

  scan_kernel<<<1, 256, 0, stream>>>(z, h0, W_Vh, W_Uh, b, out);
}
